// CausalSelfAttention_5248450036124
// MI455X (gfx1250) — matrix operand order repaired and checked by reading, not compiled, not run on hardware
//
#include <hip/hip_runtime.h>
#include <hip/hip_bf16.h>

// ---------------------------------------------------------------------------
// CDNA5 (gfx1250) wave32 WMMA attention pipeline.
// D = A * B^T GEMMs via v_wmma_f32_16x16x32_bf16, flash attention with
// windowed causal mask, all WMMA fragments loaded per the CDNA5 ISA layout
// tables (A: 8+8 split K chunks per lane; B: same 8+8 split K chunks per lane).
// ---------------------------------------------------------------------------

typedef __attribute__((ext_vector_type(16))) __bf16 bf16x16;
typedef __attribute__((ext_vector_type(8)))  float  f32x8;

__device__ __forceinline__ f32x8 zero8() {
    f32x8 z;
#pragma unroll
    for (int i = 0; i < 8; ++i) z[i] = 0.0f;
    return z;
}

__device__ __forceinline__ f32x8 wmma_bf16(bf16x16 a, bf16x16 b, f32x8 c) {
    // (neg_a, A, neg_b, B, c_mod, C, reuse_a, reuse_b)
    return __builtin_amdgcn_wmma_f32_16x16x32_bf16(false, a, false, b,
                                                   (short)0, c, false, false);
}

// A-matrix fragment (16x32 bf16, row-major, `stride` elements per row).
// Per ISA: lane (m = lane&15, g = lane>>4) holds K = 8g..8g+7 (elems 0-7)
// and K = 16+8g..16+8g+7 (elems 8-15).
__device__ __forceinline__ bf16x16 load_frag_a(const __bf16* base, int stride) {
    const int lane = threadIdx.x & 31;
    const __bf16* p = base + (lane & 15) * stride + (lane >> 4) * 8;
    union { bf16x16 v; uint4 q[2]; } u;
    u.q[0] = *reinterpret_cast<const uint4*>(p);
    u.q[1] = *reinterpret_cast<const uint4*>(p + 16);
    return u.v;
}

// B-matrix fragment. Tile stored as 16 rows (n) x 32 cols (k), row-major.
// Per ISA: lane (n = lane&15, g = lane>>4) holds K = 8g..8g+7 (elems 0-7)
// and K = 16+8g..16+8g+7 (elems 8-15).
__device__ __forceinline__ bf16x16 load_frag_b(const __bf16* base, int stride) {
    const int lane = threadIdx.x & 31;
    const __bf16* p = base + (lane & 15) * stride + (lane >> 4) * 8;
    union { bf16x16 v; uint4 q[2]; } u;
    u.q[0] = *reinterpret_cast<const uint4*>(p);
    u.q[1] = *reinterpret_cast<const uint4*>(p + 16);
    return u.v;
}

// ---------------------------------------------------------------------------
// f32 -> bf16 convert, 4 elements/thread (float4 in, 8-byte packed out)
// ---------------------------------------------------------------------------
__global__ void f32_to_bf16_kernel(const float* __restrict__ src,
                                   __bf16* __restrict__ dst, size_t n4) {
    size_t i = (size_t)blockIdx.x * blockDim.x + threadIdx.x;
    const size_t stride = (size_t)gridDim.x * blockDim.x;
    for (; i < n4; i += stride) {
        const float4 v = reinterpret_cast<const float4*>(src)[i];
        union { __bf16 e[4]; uint2 u; } o;
        o.e[0] = (__bf16)v.x; o.e[1] = (__bf16)v.y;
        o.e[2] = (__bf16)v.z; o.e[3] = (__bf16)v.w;
        reinterpret_cast<uint2*>(dst)[i] = o.u;
    }
}

// ---------------------------------------------------------------------------
// GEMM: D[M,N] = A[M,K] * Bm[N,K]^T  (both row-major, contraction contiguous)
// 128x128 block tile, 8 waves (2x4), 64x32 per wave, double-buffered LDS.
// Output type is a compile-time template parameter (straight-line epilogue).
// ---------------------------------------------------------------------------
template <bool OUT_BF16>
__global__ __launch_bounds__(256) void gemm_bf16_nt(
    const __bf16* __restrict__ A, const __bf16* __restrict__ Bm,
    float* __restrict__ Cf, __bf16* __restrict__ Cb, int M, int N, int K) {
    __shared__ __bf16 As[2][128][40];
    __shared__ __bf16 Bs[2][128][40];

    const int tid = threadIdx.x;
    const int w = tid >> 5, lane = tid & 31, g = lane >> 4, ln = lane & 15;
    const int wm = w >> 2, wn = w & 3;
    const int lrow = tid >> 1;
    const int lcol = (tid & 1) * 16;

    const __bf16* Ap = A + (size_t)(blockIdx.y * 128 + lrow) * K + lcol;
    const __bf16* Bp = Bm + (size_t)(blockIdx.x * 128 + lrow) * K + lcol;

    f32x8 acc[4][2];
#pragma unroll
    for (int i = 0; i < 4; ++i)
#pragma unroll
        for (int j = 0; j < 2; ++j) acc[i][j] = zero8();

    const int nk = K >> 5;

    uint4 a0 = *(const uint4*)(Ap);
    uint4 a1 = *(const uint4*)(Ap + 8);
    uint4 b0 = *(const uint4*)(Bp);
    uint4 b1 = *(const uint4*)(Bp + 8);
    *(uint4*)&As[0][lrow][lcol] = a0; *(uint4*)&As[0][lrow][lcol + 8] = a1;
    *(uint4*)&Bs[0][lrow][lcol] = b0; *(uint4*)&Bs[0][lrow][lcol + 8] = b1;
    __syncthreads();

    for (int kt = 0; kt < nk; ++kt) {
        const int cur = kt & 1;
        const bool more = (kt + 1 < nk);
        if (more) {
            const int kb = (kt + 1) << 5;
            a0 = *(const uint4*)(Ap + kb);
            a1 = *(const uint4*)(Ap + kb + 8);
            b0 = *(const uint4*)(Bp + kb);
            b1 = *(const uint4*)(Bp + kb + 8);
            if (kt + 2 < nk) {
                __builtin_prefetch(Ap + ((kt + 2) << 5), 0, 1);
                __builtin_prefetch(Bp + ((kt + 2) << 5), 0, 1);
            }
        }
        bf16x16 af[4], bfr[2];
#pragma unroll
        for (int i = 0; i < 4; ++i)
            af[i] = load_frag_a(&As[cur][wm * 64 + i * 16][0], 40);
#pragma unroll
        for (int j = 0; j < 2; ++j)
            bfr[j] = load_frag_b(&Bs[cur][wn * 32 + j * 16][0], 40);
#pragma unroll
        for (int i = 0; i < 4; ++i)
#pragma unroll
            for (int j = 0; j < 2; ++j)
                acc[i][j] = wmma_bf16(af[i], bfr[j], acc[i][j]);
        if (more) {
            *(uint4*)&As[cur ^ 1][lrow][lcol]     = a0;
            *(uint4*)&As[cur ^ 1][lrow][lcol + 8] = a1;
            *(uint4*)&Bs[cur ^ 1][lrow][lcol]     = b0;
            *(uint4*)&Bs[cur ^ 1][lrow][lcol + 8] = b1;
        }
        __syncthreads();
    }

    // D layout: vgpr r -> row = r + 8g, col = lane&15 (per 16x16 tile)
#pragma unroll
    for (int i = 0; i < 4; ++i) {
        const int rbase = blockIdx.y * 128 + wm * 64 + i * 16 + 8 * g;
#pragma unroll
        for (int j = 0; j < 2; ++j) {
            const int cc = blockIdx.x * 128 + wn * 32 + j * 16 + ln;
#pragma unroll
            for (int r = 0; r < 8; ++r) {
                const size_t idx = (size_t)(rbase + r) * N + cc;
                if constexpr (OUT_BF16) Cb[idx] = (__bf16)acc[i][j][r];
                else                    Cf[idx] = acc[i][j][r];
            }
        }
    }
}

// ---------------------------------------------------------------------------
// Pointwise: gate + v-update, RoPE + RMS-norm * 1.2 on q/k.
// One block per token; 16 lanes per head (h = tid>>4, j = tid&15).
// Outputs in [B][H][T][128] bf16 for the attention kernel.
// ---------------------------------------------------------------------------
__global__ __launch_bounds__(256) void pointwise_kernel(
    const __bf16* __restrict__ q_raw, const __bf16* __restrict__ k_raw,
    const __bf16* __restrict__ v_raw, const float* __restrict__ x,
    const float* __restrict__ ve, const float* __restrict__ Wg,
    const float* __restrict__ cosp, const float* __restrict__ sinp,
    __bf16* __restrict__ qh, __bf16* __restrict__ kh,
    __bf16* __restrict__ vh, int T) {
    const int tok = blockIdx.x;        // b*T + t
    const int t = tok % T;
    const int b = tok / T;
    const int h = threadIdx.x >> 4;
    const int j = threadIdx.x & 15;
    const size_t rowoff = (size_t)tok * 2048 + h * 128;
    const size_t outoff = ((size_t)(b * 16 + h) * T + t) * 128;

    // gate = 3 * sigmoid(x[:12] . Wg[h])
    float dg = 0.0f;
#pragma unroll
    for (int i = 0; i < 12; ++i) dg += x[(size_t)tok * 2048 + i] * Wg[h * 12 + i];
    const float gate = 3.0f / (1.0f + __expf(-dg));

    // v = v_raw + gate * ve
#pragma unroll
    for (int e = 0; e < 8; ++e) {
        const int d = j * 8 + e;
        const float vv = (float)v_raw[rowoff + d] + gate * ve[rowoff + d];
        vh[outoff + d] = (__bf16)vv;
    }

    // q: rope + rmsnorm * 1.2
    {
        float o1[4], o2[4], ss = 0.0f;
#pragma unroll
        for (int e = 0; e < 4; ++e) {
            const int p = j * 4 + e;
            const float c = cosp[t * 64 + p], s = sinp[t * 64 + p];
            const float x1 = (float)q_raw[rowoff + p];
            const float x2 = (float)q_raw[rowoff + p + 64];
            o1[e] = x1 * c + x2 * s;
            o2[e] = -x1 * s + x2 * c;
            ss += o1[e] * o1[e] + o2[e] * o2[e];
        }
#pragma unroll
        for (int m = 1; m < 16; m <<= 1) ss += __shfl_xor(ss, m, 32);
        const float rn = rsqrtf(ss * (1.0f / 128.0f) + 1e-6f) * 1.2f;
#pragma unroll
        for (int e = 0; e < 4; ++e) {
            const int p = j * 4 + e;
            qh[outoff + p]      = (__bf16)(o1[e] * rn);
            qh[outoff + p + 64] = (__bf16)(o2[e] * rn);
        }
    }
    // k: rope + rmsnorm * 1.2
    {
        float o1[4], o2[4], ss = 0.0f;
#pragma unroll
        for (int e = 0; e < 4; ++e) {
            const int p = j * 4 + e;
            const float c = cosp[t * 64 + p], s = sinp[t * 64 + p];
            const float x1 = (float)k_raw[rowoff + p];
            const float x2 = (float)k_raw[rowoff + p + 64];
            o1[e] = x1 * c + x2 * s;
            o2[e] = -x1 * s + x2 * c;
            ss += o1[e] * o1[e] + o2[e] * o2[e];
        }
#pragma unroll
        for (int m = 1; m < 16; m <<= 1) ss += __shfl_xor(ss, m, 32);
        const float rn = rsqrtf(ss * (1.0f / 128.0f) + 1e-6f) * 1.2f;
#pragma unroll
        for (int e = 0; e < 4; ++e) {
            const int p = j * 4 + e;
            kh[outoff + p]      = (__bf16)(o1[e] * rn);
            kh[outoff + p + 64] = (__bf16)(o2[e] * rn);
        }
    }
}

// ---------------------------------------------------------------------------
// Flash attention, windowed causal. Block = (b, h, 128-query tile); 8 waves,
// 16 query rows per wave; key tiles of 32 streamed through LDS.
// y written as [B*T, H*128] bf16 (row-major) for the projection GEMM.
// ---------------------------------------------------------------------------
__global__ __launch_bounds__(256) void attn_kernel(
    const __bf16* __restrict__ qh, const __bf16* __restrict__ kh,
    const __bf16* __restrict__ vh, __bf16* __restrict__ y,
    const int* __restrict__ wptr, int T) {
    __shared__ __bf16 Ks[32][136];     // K tile  [n][d]
    __shared__ __bf16 Vt[128][40];     // V tile transposed: [d][n]
    __shared__ __bf16 Ps[8][16][40];   // per-wave P scratch (A-layout staging)

    const int b = blockIdx.z, h = blockIdx.y;
    const int q0 = blockIdx.x * 128;
    const int tid = threadIdx.x;
    const int w = tid >> 5, lane = tid & 31, g = lane >> 4, ln = lane & 15;
    const size_t bh = ((size_t)(b * 16 + h)) * T * 128;

    const int win = *wptr;
    const bool use_win = (win > 0 && win < T);
    const float SCALE = 0.08838834764831845f;  // 1/sqrt(128)

    // Preload Q A-fragments for this wave's 16 rows (global layout matches).
    bf16x16 aq[4];
#pragma unroll
    for (int kt = 0; kt < 4; ++kt)
        aq[kt] = load_frag_a(qh + bh + (size_t)(q0 + w * 16) * 128 + kt * 32, 128);

    f32x8 o[8];
    float mrow[8], lrow[8];
#pragma unroll
    for (int r = 0; r < 8; ++r) { mrow[r] = -1e30f; lrow[r] = 0.0f; }
#pragma unroll
    for (int dt = 0; dt < 8; ++dt) o[dt] = zero8();

    int klo = 0;
    if (use_win) { klo = q0 - win; if (klo < 0) klo = 0; klo &= ~31; }

    const int kvrow = tid >> 3;        // 0..31
    const int kvcol = (tid & 7) * 16;  // 0,16,..,112

    for (int kb = klo; kb < q0 + 128; kb += 32) {
        __syncthreads();
        // Load K tile and transposed V tile.
        {
            const __bf16* ksrc = kh + bh + (size_t)(kb + kvrow) * 128 + kvcol;
            *(uint4*)&Ks[kvrow][kvcol]     = *(const uint4*)ksrc;
            *(uint4*)&Ks[kvrow][kvcol + 8] = *(const uint4*)(ksrc + 8);
            const __bf16* vsrc = vh + bh + (size_t)(kb + kvrow) * 128 + kvcol;
            union { uint4 u[2]; __bf16 e[16]; } vv;
            vv.u[0] = *(const uint4*)vsrc;
            vv.u[1] = *(const uint4*)(vsrc + 8);
#pragma unroll
            for (int e2 = 0; e2 < 16; ++e2) Vt[kvcol + e2][kvrow] = vv.e[e2];
        }
        __syncthreads();

        // S = Q * K^T  (two 16-column groups)
        f32x8 s0 = zero8(), s1 = zero8();
#pragma unroll
        for (int kt = 0; kt < 4; ++kt) {
            bf16x16 bk0 = load_frag_b(&Ks[0][kt * 32], 136);
            bf16x16 bk1 = load_frag_b(&Ks[16][kt * 32], 136);
            s0 = wmma_bf16(aq[kt], bk0, s0);
            s1 = wmma_bf16(aq[kt], bk1, s1);
        }

        // Online softmax + stage P into LDS (bf16, A layout staging).
        const int n0 = kb + ln, n1 = n0 + 16;
#pragma unroll
        for (int r = 0; r < 8; ++r) {
            const int qrow = q0 + w * 16 + r + 8 * g;
            float x0 = (qrow >= n0 && (!use_win || qrow - n0 <= win))
                           ? s0[r] * SCALE : -1e9f;
            float x1 = (qrow >= n1 && (!use_win || qrow - n1 <= win))
                           ? s1[r] * SCALE : -1e9f;
            float mx = fmaxf(x0, x1);
#pragma unroll
            for (int m = 1; m < 16; m <<= 1) mx = fmaxf(mx, __shfl_xor(mx, m, 32));
            const float mnew = fmaxf(mrow[r], mx);
            const float p0 = __expf(x0 - mnew);
            const float p1 = __expf(x1 - mnew);
            const float corr = __expf(mrow[r] - mnew);
            float ps = p0 + p1;
#pragma unroll
            for (int m = 1; m < 16; m <<= 1) ps += __shfl_xor(ps, m, 32);
            lrow[r] = lrow[r] * corr + ps;
            mrow[r] = mnew;
#pragma unroll
            for (int dt = 0; dt < 8; ++dt) o[dt][r] *= corr;
            Ps[w][r + 8 * g][ln]      = (__bf16)p0;
            Ps[w][r + 8 * g][ln + 16] = (__bf16)p1;
        }

        // O += P * V   (P: 16x32 A-frag; V: B-frags from transposed tile)
        bf16x16 ap = load_frag_a(&Ps[w][0][0], 40);
#pragma unroll
        for (int dt = 0; dt < 8; ++dt) {
            bf16x16 bv = load_frag_b(&Vt[dt * 16][0], 40);
            o[dt] = wmma_bf16(ap, bv, o[dt]);
        }
    }

    // Epilogue: y[b*T+q][h*128+d] = O / l
#pragma unroll
    for (int r = 0; r < 8; ++r) {
        const int qrow = q0 + w * 16 + r + 8 * g;
        const float inv = 1.0f / lrow[r];
        const size_t base = ((size_t)(b * T + qrow)) * 2048 + h * 128;
#pragma unroll
        for (int dt = 0; dt < 8; ++dt)
            y[base + dt * 16 + ln] = (__bf16)(o[dt][r] * inv);
    }
}

// ---------------------------------------------------------------------------
// Host side
// ---------------------------------------------------------------------------
extern "C" void kernel_launch(void* const* d_in, const int* in_sizes, int n_in,
                              void* d_out, int out_size, void* d_ws,
                              size_t ws_size, hipStream_t stream) {
    (void)n_in; (void)out_size; (void)ws_size;
    const float* x    = (const float*)d_in[0];
    const float* ve   = (const float*)d_in[1];
    const float* Wq   = (const float*)d_in[2];
    const float* Wk   = (const float*)d_in[3];
    const float* Wv   = (const float*)d_in[4];
    const float* Wp   = (const float*)d_in[5];
    const float* Wg   = (const float*)d_in[6];
    const float* cosp = (const float*)d_in[7];
    const float* sinp = (const float*)d_in[8];
    const int*   wptr = (const int*)d_in[9];

    const int C = 2048, H = 16, D = 128;
    const int T = in_sizes[7] / (D / 2);       // cos: (T,1,64)
    const int B = (int)(in_sizes[0] / ((size_t)T * C));
    const size_t M = (size_t)B * T;

    char* ws = (char*)d_ws;
    size_t off = 0;
    auto alloc = [&](size_t bytes) -> void* {
        void* p = ws + off;
        off = (off + bytes + 255) & ~(size_t)255;
        return p;
    };
    __bf16* xb    = (__bf16*)alloc(M * C * 2);
    __bf16* Wqb   = (__bf16*)alloc((size_t)C * C * 2);
    __bf16* Wkb   = (__bf16*)alloc((size_t)C * C * 2);
    __bf16* Wvb   = (__bf16*)alloc((size_t)C * C * 2);
    __bf16* Wpb   = (__bf16*)alloc((size_t)C * C * 2);
    __bf16* q_raw = (__bf16*)alloc(M * C * 2);
    __bf16* k_raw = (__bf16*)alloc(M * C * 2);
    __bf16* v_raw = (__bf16*)alloc(M * C * 2);
    __bf16* qhb   = (__bf16*)alloc(M * C * 2);
    __bf16* khb   = (__bf16*)alloc(M * C * 2);
    __bf16* vhb   = (__bf16*)alloc(M * C * 2);
    __bf16* yb    = (__bf16*)alloc(M * C * 2);

    // 1. converts (element counts are multiples of 4)
    f32_to_bf16_kernel<<<1024, 256, 0, stream>>>(x, xb, M * C / 4);
    f32_to_bf16_kernel<<<512, 256, 0, stream>>>(Wq, Wqb, (size_t)C * C / 4);
    f32_to_bf16_kernel<<<512, 256, 0, stream>>>(Wk, Wkb, (size_t)C * C / 4);
    f32_to_bf16_kernel<<<512, 256, 0, stream>>>(Wv, Wvb, (size_t)C * C / 4);
    f32_to_bf16_kernel<<<512, 256, 0, stream>>>(Wp, Wpb, (size_t)C * C / 4);

    // 2. QKV projections (WMMA GEMMs, bf16 out)
    dim3 ggrid(C / 128, (unsigned)(M / 128));
    gemm_bf16_nt<true><<<ggrid, 256, 0, stream>>>(xb, Wqb, nullptr, q_raw,
                                                  (int)M, C, C);
    gemm_bf16_nt<true><<<ggrid, 256, 0, stream>>>(xb, Wkb, nullptr, k_raw,
                                                  (int)M, C, C);
    gemm_bf16_nt<true><<<ggrid, 256, 0, stream>>>(xb, Wvb, nullptr, v_raw,
                                                  (int)M, C, C);

    // 3. gate + rope + rmsnorm, relayout to [B,H,T,D]
    pointwise_kernel<<<(unsigned)M, 256, 0, stream>>>(
        q_raw, k_raw, v_raw, x, ve, Wg, cosp, sinp, qhb, khb, vhb, T);

    // 4. flash attention (WMMA)
    attn_kernel<<<dim3(T / 128, H, B), 256, 0, stream>>>(qhb, khb, vhb, yb,
                                                         wptr, T);

    // 5. output projection (WMMA GEMM, f32 out)
    gemm_bf16_nt<false><<<ggrid, 256, 0, stream>>>(yb, Wpb, (float*)d_out,
                                                   nullptr, (int)M, C, C);
}
